// RNNModel_55009941127467
// MI455X (gfx1250) — matrix operand order repaired and checked by reading, not compiled, not run on hardware
//
#include <hip/hip_runtime.h>
#include <cstdint>
#include <cstddef>

// ---------------- problem constants ----------------
#define TT 2048
#define BB 64
#define II 256
#define HH 512
// gate row width = 3*HH (r,i gates = 2H from W_ih, then H from W_in)

// ---------------- vector typedefs ----------------
typedef __bf16 bf16;
typedef bf16  bf16x8  __attribute__((ext_vector_type(8)));
typedef bf16  bf16x16 __attribute__((ext_vector_type(16)));
typedef float f32x8   __attribute__((ext_vector_type(8)));
typedef unsigned int u32x4 __attribute__((ext_vector_type(4)));
typedef int   i32x4   __attribute__((ext_vector_type(4)));
typedef int   i32x8   __attribute__((ext_vector_type(8)));

// ---------------- TDM: 2D bf16 tile -> LDS ----------------
// Builds the D# descriptor (groups 0/1; groups 2/3 zero for 2D) per
// CDNA5 ISA and issues tensor_load_to_lds (6-arg
// clang-23 form: g0, g1, g2, g3, extra group, cpol).
// Pad feature: after every 16 DWORDs (32 bf16 = one tile row) insert 4 DWORDs
// (8 bf16) => LDS row stride 40 bf16, conflict-free for the WMMA frag reads.
__device__ __forceinline__ void tdm_load_2d_bf16(unsigned lds_off, const bf16* gptr,
                                                 unsigned tile_w, unsigned tile_h,
                                                 unsigned long long stride_elems,
                                                 unsigned tensor_w, unsigned tensor_h) {
  unsigned long long ga = (unsigned long long)(uintptr_t)gptr;
  u32x4 g0;
  g0[0] = 1u;                                            // count=1 (valid user D#)
  g0[1] = lds_off;                                       // lds_addr (bytes)
  g0[2] = (unsigned)(ga & 0xffffffffu);                  // global_addr[31:0]
  g0[3] = (unsigned)((ga >> 32) & 0x01ffffffu) | (2u << 30); // addr[56:32] | type=2
  i32x8 g1;
  unsigned w0 = (1u << 16)                               // data_size = 2 bytes
              | (1u << 20) | (3u << 22) | (3u << 25);    // pad_en, interval=16dw, amount=4dw
  g1[0] = (int)w0;
  g1[1] = (int)((tensor_w & 0xffffu) << 16);             // tensor_dim0[15:0] @ bits 63:48
  g1[2] = (int)(((tensor_w >> 16) & 0xffffu) | ((tensor_h & 0xffffu) << 16));
  g1[3] = (int)(((tensor_h >> 16) & 0xffffu) | ((tile_w & 0xffffu) << 16));
  g1[4] = (int)(tile_h & 0xffffu);                       // tile_dim1 (tile_dim2=0)
  g1[5] = (int)(stride_elems & 0xffffffffu);             // tensor_dim0_stride[31:0]
  g1[6] = (int)((stride_elems >> 32) & 0xffffu);         // stride[47:32]
  g1[7] = 0;
  i32x4 z4 = {0, 0, 0, 0};
  i32x8 z8 = {0, 0, 0, 0, 0, 0, 0, 0};
  __builtin_amdgcn_tensor_load_to_lds(g0, g1, z4, z4, z8, 0);
}

// ---------------- WMMA fragment loaders (ISA §7.12.2 layouts) ----------------
// A (16x32 bf16, MxK): lane L holds row M=L&15; lanes<16: K in [0,8)+[16,24),
// lanes>=16: K in [8,16)+[24,32) => two contiguous 8-elem runs.
__device__ __forceinline__ bf16x16 frag_a(const bf16* row, int lane) {
  int koff = (lane >> 4) * 8;
  bf16x8 a = *(const bf16x8*)(row + koff);
  bf16x8 b = *(const bf16x8*)(row + 16 + koff);
  return __builtin_shufflevector(a, b, 0,1,2,3,4,5,6,7,8,9,10,11,12,13,14,15);
}
// B (32x16 bf16, KxN) stored in LDS as [n][k] (ldr elems): lane L holds col
// N=L&15; lanes<16: K in [0,8)+[16,24), lanes>=16: K in [8,16)+[24,32).
__device__ __forceinline__ bf16x16 frag_b(const bf16* base, int ldr, int lane) {
  const bf16* row = base + (size_t)(lane & 15) * ldr + ((lane >> 4) * 8);
  bf16x8 a = *(const bf16x8*)(row);
  bf16x8 b = *(const bf16x8*)(row + 16);
  return __builtin_shufflevector(a, b, 0,1,2,3,4,5,6,7,8,9,10,11,12,13,14,15);
}

__device__ __forceinline__ f32x8 wmma_bf16(bf16x16 a, bf16x16 b, f32x8 c) {
  return __builtin_amdgcn_wmma_f32_16x16x32_bf16(false, a, false, b, (short)0, c, false, false);
}

__device__ __forceinline__ float sigmoidf(float x) { return 1.0f / (1.0f + __expf(-x)); }

// ---------------- utility kernels ----------------
__global__ void zero_f32_kernel(float* p, size_t n) {
  for (size_t i = blockIdx.x * blockDim.x + threadIdx.x; i < n; i += (size_t)gridDim.x * blockDim.x)
    p[i] = 0.0f;
}

// fp32 -> (hi, lo) bf16 split: x ~= hi + lo with ~16 mantissa bits total
__global__ void split_bf16_kernel(const float* __restrict__ src, bf16* __restrict__ hi,
                                  bf16* __restrict__ lo, size_t n) {
  for (size_t i = blockIdx.x * blockDim.x + threadIdx.x; i < n; i += (size_t)gridDim.x * blockDim.x) {
    float v = src[i];
    bf16 h = (bf16)v;
    hi[i] = h;
    lo[i] = (bf16)(v - (float)h);
  }
}

// final h_n assembly: out[T*B*H ..] = [h0_final ; out1[T-1]]
__global__ void tail_kernel(const float* __restrict__ h0, float* __restrict__ out) {
  const size_t BH = (size_t)BB * HH;
  for (size_t i = blockIdx.x * blockDim.x + threadIdx.x; i < BH; i += (size_t)gridDim.x * blockDim.x) {
    out[(size_t)TT * BH + i] = h0[i];
    out[(size_t)TT * BH + BH + i] = out[(size_t)(TT - 1) * BH + i];
  }
}

// ---------------- big split-bf16 GEMM: C = A * B (+bias) ----------------
// A: M x K, row-major bf16 hi/lo planes.  TRANSB: B stored N x K row-major
// (out[m,n] = sum_k A[m,k]*B[n,k]); else B stored K x N row-major.
// C: f32, leading dim ldc, column offset coloff.  64x64 tile / 8 waves.
template<bool TRANSB>
__global__ __launch_bounds__(256)
void gemm_split_kernel(const bf16* __restrict__ Ah, const bf16* __restrict__ Al,
                       const bf16* __restrict__ Bh, const bf16* __restrict__ Bl,
                       const float* __restrict__ bias, float* __restrict__ C,
                       int M, int K, int ldb, int ldc, int coloff) {
  __shared__ bf16 As_h[64][40];
  __shared__ bf16 As_l[64][40];
  __shared__ bf16 Bs_h[64][40];  // [n][k]
  __shared__ bf16 Bs_l[64][40];
  const int tid = threadIdx.x, lane = tid & 31, wid = tid >> 5;
  const int mt = wid & 3, np = wid >> 2;           // wave -> (m-tile, n-pair)
  const int m0 = blockIdx.y * 64, n0 = blockIdx.x * 64;
  f32x8 acc0 = {0.f,0.f,0.f,0.f,0.f,0.f,0.f,0.f};
  f32x8 acc1 = acc0;

  for (int k0 = 0; k0 < K; k0 += 32) {
    if (TRANSB) {  // rows are n, contiguous k: vector copy
      int n = tid >> 2, kc = (tid & 3) * 8;
      *(bf16x8*)&Bs_h[n][kc] = *(const bf16x8*)&Bh[(size_t)(n0 + n) * ldb + k0 + kc];
      *(bf16x8*)&Bs_l[n][kc] = *(const bf16x8*)&Bl[(size_t)(n0 + n) * ldb + k0 + kc];
    } else {       // rows are k, contiguous n: transpose into [n][k]
      int k = tid >> 3, nc = (tid & 7) * 8;
#pragma unroll
      for (int j = 0; j < 8; ++j) {
        Bs_h[nc + j][k] = Bh[(size_t)(k0 + k) * ldb + n0 + nc + j];
        Bs_l[nc + j][k] = Bl[(size_t)(k0 + k) * ldb + n0 + nc + j];
      }
    }
    if (wid == 0) {  // one wave drives the TDM for both A planes
      tdm_load_2d_bf16((unsigned)(uintptr_t)&As_h[0][0], Ah + (size_t)m0 * K + k0,
                       32, 64, (unsigned long long)K, (unsigned)K, (unsigned)M);
      tdm_load_2d_bf16((unsigned)(uintptr_t)&As_l[0][0], Al + (size_t)m0 * K + k0,
                       32, 64, (unsigned long long)K, (unsigned)K, (unsigned)M);
      __builtin_amdgcn_s_wait_tensorcnt(0);
    }
    __syncthreads();

    bf16x16 ah = frag_a(&As_h[mt * 16 + (lane & 15)][0], lane);
    bf16x16 al = frag_a(&As_l[mt * 16 + (lane & 15)][0], lane);
    {
      bf16x16 bh = frag_b(&Bs_h[np * 32][0], 40, lane);
      bf16x16 bl = frag_b(&Bs_l[np * 32][0], 40, lane);
      acc0 = wmma_bf16(ah, bh, acc0);
      acc0 = wmma_bf16(ah, bl, acc0);   // split-bf16 cross terms
      acc0 = wmma_bf16(al, bh, acc0);
    }
    {
      bf16x16 bh = frag_b(&Bs_h[np * 32 + 16][0], 40, lane);
      bf16x16 bl = frag_b(&Bs_l[np * 32 + 16][0], 40, lane);
      acc1 = wmma_bf16(ah, bh, acc1);
      acc1 = wmma_bf16(ah, bl, acc1);
      acc1 = wmma_bf16(al, bh, acc1);
    }
    __syncthreads();
  }

#pragma unroll
  for (int e = 0; e < 8; ++e) {   // C layout: M = e + 8*(lane>=16), N = lane&15
    int r   = m0 + mt * 16 + (lane >> 4) * 8 + e;
    int c0g = n0 + np * 32 + (lane & 15);
    int c1g = c0g + 16;
    float b0 = bias ? bias[c0g] : 0.0f;
    float b1 = bias ? bias[c1g] : 0.0f;
    C[(size_t)r * ldc + coloff + c0g] = acc0[e] + b0;
    C[(size_t)r * ldc + coloff + c1g] = acc1[e] + b1;
  }
}

// ---------------- recurrent GRU step (one launch per t, per layer) ----------------
// Grid: H/16 = 32 WGs of 128 threads (4 waves, one 16-row m-tile each).
// Each WG covers 16 output columns; computes three K=512 GEMM columns sets
// (W_hh rows n, W_hh rows H+n, W_hn cols n) with split-bf16 WMMA, then gates.
__global__ __launch_bounds__(128)
void gru_step_kernel(const bf16* __restrict__ hAh, const bf16* __restrict__ hAl,
                     const float* __restrict__ hinF,
                     const bf16* __restrict__ Whh_h, const bf16* __restrict__ Whh_l,
                     const float* __restrict__ b_hh,
                     const bf16* __restrict__ Whn_h, const bf16* __restrict__ Whn_l,
                     const float* __restrict__ grow, const float* __restrict__ bC,
                     float* __restrict__ houtF,
                     bf16* __restrict__ houtH, bf16* __restrict__ houtL) {
  __shared__ bf16 As_h[64][40];
  __shared__ bf16 As_l[64][40];
  __shared__ bf16 Bs[3][2][16][40];   // [mat r/i/n][hi,lo][n][k]
  const int tid = threadIdx.x, lane = tid & 31, wid = tid >> 5;
  const int n0 = blockIdx.x * 16;
  f32x8 zf = {0.f,0.f,0.f,0.f,0.f,0.f,0.f,0.f};
  f32x8 acc[3] = {zf, zf, zf};

  for (int k0 = 0; k0 < HH; k0 += 32) {
    {  // W_hh r-rows (n0..) and i-rows (HH+n0..), trans layout: contiguous K
      int half = tid >> 6, tt = tid & 63;
      int n = tt >> 2, kc = (tt & 3) * 8;
      int rrow = half * HH + n0 + n;
      *(bf16x8*)&Bs[half][0][n][kc] = *(const bf16x8*)&Whh_h[(size_t)rrow * HH + k0 + kc];
      *(bf16x8*)&Bs[half][1][n][kc] = *(const bf16x8*)&Whh_l[(size_t)rrow * HH + k0 + kc];
    }
    {  // W_hn (K x N row-major): transpose into [n][k]
      int k = tid >> 2, nc = (tid & 3) * 4;
#pragma unroll
      for (int j = 0; j < 4; ++j) {
        Bs[2][0][nc + j][k] = Whn_h[(size_t)(k0 + k) * HH + n0 + nc + j];
        Bs[2][1][nc + j][k] = Whn_l[(size_t)(k0 + k) * HH + n0 + nc + j];
      }
    }
    if (wid == 0) {  // TDM: h (64 x 512) tile hi/lo
      tdm_load_2d_bf16((unsigned)(uintptr_t)&As_h[0][0], hAh + k0, 32, 64, HH, HH, BB);
      tdm_load_2d_bf16((unsigned)(uintptr_t)&As_l[0][0], hAl + k0, 32, 64, HH, HH, BB);
      __builtin_amdgcn_s_wait_tensorcnt(0);
    }
    __syncthreads();

    bf16x16 ah = frag_a(&As_h[wid * 16 + (lane & 15)][0], lane);
    bf16x16 al = frag_a(&As_l[wid * 16 + (lane & 15)][0], lane);
#pragma unroll
    for (int m = 0; m < 3; ++m) {
      bf16x16 bh = frag_b(&Bs[m][0][0][0], 40, lane);
      bf16x16 bl = frag_b(&Bs[m][1][0][0], 40, lane);
      acc[m] = wmma_bf16(ah, bh, acc[m]);
      acc[m] = wmma_bf16(ah, bl, acc[m]);
      acc[m] = wmma_bf16(al, bh, acc[m]);
    }
    __syncthreads();
  }

  const int col = n0 + (lane & 15);
  const float bhr = b_hh[col], bhi = b_hh[HH + col], bc = bC[col];
#pragma unroll
  for (int e = 0; e < 8; ++e) {
    int r = wid * 16 + (lane >> 4) * 8 + e;
    const float* gr = grow + (size_t)r * (3 * HH);
    float i_r = gr[col], i_i = gr[HH + col], i_n = gr[2 * HH + col];
    float reset = sigmoidf(i_r + acc[0][e] + bhr);
    float ig    = sigmoidf(i_i + acc[1][e] + bhi);
    float cand  = i_n + reset * acc[2][e];
    float sg = (cand > 0.f) ? 1.f : ((cand < 0.f) ? -1.f : 0.f);
    float ng = fmaxf(fabsf(cand) + bc, 0.f) * sg;
    float hold = hinF[(size_t)r * HH + col];
    float hy = ng + ig * (hold - ng);
    houtF[(size_t)r * HH + col] = hy;
    bf16 hh = (bf16)hy;
    houtH[(size_t)r * HH + col] = hh;
    houtL[(size_t)r * HH + col] = (bf16)(hy - (float)hh);
  }
}

// ---------------- host orchestration ----------------
extern "C" void kernel_launch(void* const* d_in, const int* in_sizes, int n_in,
                              void* d_out, int out_size, void* d_ws, size_t ws_size,
                              hipStream_t stream) {
  (void)in_sizes; (void)n_in; (void)out_size; (void)ws_size;
  const float* x    = (const float*)d_in[0];
  const float* Wih0 = (const float*)d_in[1];
  const float* bih0 = (const float*)d_in[2];
  const float* Whh0 = (const float*)d_in[3];
  const float* bhh0 = (const float*)d_in[4];
  const float* Win0 = (const float*)d_in[5];
  const float* Whn0 = (const float*)d_in[6];
  const float* bC0  = (const float*)d_in[7];
  const float* Wih1 = (const float*)d_in[8];
  const float* bih1 = (const float*)d_in[9];
  const float* Whh1 = (const float*)d_in[10];
  const float* bhh1 = (const float*)d_in[11];
  const float* Win1 = (const float*)d_in[12];
  const float* Whn1 = (const float*)d_in[13];
  const float* bC1  = (const float*)d_in[14];
  float* out = (float*)d_out;

  const size_t TB  = (size_t)TT * BB;       // 131072 rows
  const size_t TBI = TB * II;               // x elems
  const size_t TBH = TB * HH;               // out0 elems
  const size_t BH  = (size_t)BB * HH;

  char* ws = (char*)d_ws;
  size_t off = 0;
  auto alloc = [&](size_t bytes) -> void* {
    void* p = ws + off;
    off = (off + bytes + 255) & ~(size_t)255;
    return p;
  };

  bf16*  xs_h = (bf16*)alloc(TBI * 2);
  bf16*  xs_l = (bf16*)alloc(TBI * 2);
  bf16*  o0_h = (bf16*)alloc(TBH * 2);
  bf16*  o0_l = (bf16*)alloc(TBH * 2);
  float* gate = (float*)alloc(TB * (3 * HH) * 4);   // reused for both layers
  // weight hi/lo planes
  auto wplanes = [&](size_t n, bf16*& h, bf16*& l) { h = (bf16*)alloc(n * 2); l = (bf16*)alloc(n * 2); };
  bf16 *wih0h,*wih0l,*whh0h,*whh0l,*win0h,*win0l,*whn0h,*whn0l;
  bf16 *wih1h,*wih1l,*whh1h,*whh1l,*win1h,*win1l,*whn1h,*whn1l;
  wplanes((size_t)2*HH*II, wih0h, wih0l);
  wplanes((size_t)2*HH*HH, whh0h, whh0l);
  wplanes((size_t)II*HH,   win0h, win0l);
  wplanes((size_t)HH*HH,   whn0h, whn0l);
  wplanes((size_t)2*HH*HH, wih1h, wih1l);
  wplanes((size_t)2*HH*HH, whh1h, whh1l);
  wplanes((size_t)HH*HH,   win1h, win1l);
  wplanes((size_t)HH*HH,   whn1h, whn1l);
  float* hb[2] = { (float*)alloc(BH * 4), (float*)alloc(BH * 4) };  // layer-0 h ping-pong
  bf16*  s1h[2] = { (bf16*)alloc(BH * 2), (bf16*)alloc(BH * 2) };   // layer-1 h split ping-pong
  bf16*  s1l[2] = { (bf16*)alloc(BH * 2), (bf16*)alloc(BH * 2) };
  float* zeroF = (float*)alloc(BH * 4);   // zeros as f32 AND as bf16 planes
  bf16*  zeroB = (bf16*)zeroF;

  // --- init (re-done every call: deterministic) ---
  zero_f32_kernel<<<64, 256, 0, stream>>>(zeroF, BH);
  zero_f32_kernel<<<64, 256, 0, stream>>>(hb[0], BH);

  // --- split inputs/weights to bf16 hi/lo ---
  split_bf16_kernel<<<2048, 256, 0, stream>>>(x, xs_h, xs_l, TBI);
  split_bf16_kernel<<<256, 256, 0, stream>>>(Wih0, wih0h, wih0l, (size_t)2*HH*II);
  split_bf16_kernel<<<256, 256, 0, stream>>>(Whh0, whh0h, whh0l, (size_t)2*HH*HH);
  split_bf16_kernel<<<256, 256, 0, stream>>>(Win0, win0h, win0l, (size_t)II*HH);
  split_bf16_kernel<<<256, 256, 0, stream>>>(Whn0, whn0h, whn0l, (size_t)HH*HH);
  split_bf16_kernel<<<256, 256, 0, stream>>>(Wih1, wih1h, wih1l, (size_t)2*HH*HH);
  split_bf16_kernel<<<256, 256, 0, stream>>>(Whh1, whh1h, whh1l, (size_t)2*HH*HH);
  split_bf16_kernel<<<256, 256, 0, stream>>>(Win1, win1h, win1l, (size_t)HH*HH);
  split_bf16_kernel<<<256, 256, 0, stream>>>(Whn1, whn1h, whn1l, (size_t)HH*HH);

  const int M = (int)TB;
  // --- layer 0 pre-GEMMs: gate[:, 0:1024] = x@W_ih0^T + b ; gate[:, 1024:1536] = x@W_in0 ---
  gemm_split_kernel<true ><<<dim3(2 * HH / 64, M / 64), 256, 0, stream>>>(
      xs_h, xs_l, wih0h, wih0l, bih0, gate, M, II, II, 3 * HH, 0);
  gemm_split_kernel<false><<<dim3(HH / 64, M / 64), 256, 0, stream>>>(
      xs_h, xs_l, win0h, win0l, nullptr, gate, M, II, HH, 3 * HH, 2 * HH);

  // --- layer 0 scan ---
  for (int t = 0; t < TT; ++t) {
    const bf16* Ah  = (t == 0) ? zeroB : o0_h + (size_t)(t - 1) * BH;
    const bf16* Al  = (t == 0) ? zeroB : o0_l + (size_t)(t - 1) * BH;
    const float* hin  = hb[t & 1];
    float*       hout = hb[(t + 1) & 1];
    const float* grow = gate + (size_t)t * BB * (3 * HH);
    gru_step_kernel<<<HH / 16, 128, 0, stream>>>(
        Ah, Al, hin, whh0h, whh0l, bhh0, whn0h, whn0l, grow, bC0,
        hout, o0_h + (size_t)t * BH, o0_l + (size_t)t * BH);
  }

  // --- layer 1 pre-GEMMs (gate buffer reused) ---
  gemm_split_kernel<true ><<<dim3(2 * HH / 64, M / 64), 256, 0, stream>>>(
      o0_h, o0_l, wih1h, wih1l, bih1, gate, M, HH, HH, 3 * HH, 0);
  gemm_split_kernel<false><<<dim3(HH / 64, M / 64), 256, 0, stream>>>(
      o0_h, o0_l, win1h, win1l, nullptr, gate, M, HH, HH, 3 * HH, 2 * HH);

  // --- layer 1 scan: out1 rows live directly in d_out (also serve as h f32) ---
  for (int t = 0; t < TT; ++t) {
    const bf16* Ah  = (t == 0) ? zeroB : s1h[t & 1];
    const bf16* Al  = (t == 0) ? zeroB : s1l[t & 1];
    const float* hin  = (t == 0) ? zeroF : out + (size_t)(t - 1) * BH;
    float*       hout = out + (size_t)t * BH;
    const float* grow = gate + (size_t)t * BB * (3 * HH);
    gru_step_kernel<<<HH / 16, 128, 0, stream>>>(
        Ah, Al, hin, whh1h, whh1l, bhh1, whn1h, whn1l, grow, bC1,
        hout, s1h[(t + 1) & 1], s1l[(t + 1) & 1]);
  }

  // --- final hidden states: [h0_final ; out1[T-1]] appended after out1 ---
  tail_kernel<<<128, 256, 0, stream>>>(hb[0], out);
}
